// PCNGuidedAttention_85246510891352
// MI455X (gfx1250) — matrix operand order repaired and checked by reading, not compiled, not run on hardware
//
#include <hip/hip_runtime.h>
#include <hip/hip_bf16.h>

// ---------------------------------------------------------------------------
// PCN guided attention for MI455X (gfx1250, wave32, WMMA).
// GEMMs: v_wmma_f32_16x16x32_f16, BK=64 tiles, double-buffered LDS fed by
// GLOBAL_LOAD_ASYNC_TO_LDS_B128 (ASYNCcnt). Attention: flash-style WMMA with
// online softmax, async-staged K tiles.
// ---------------------------------------------------------------------------

typedef _Float16 half_t;
typedef __attribute__((ext_vector_type(4)))  _Float16 v4h;
typedef __attribute__((ext_vector_type(8)))  _Float16 v8h;
typedef __attribute__((ext_vector_type(16))) _Float16 v16h;
typedef __attribute__((ext_vector_type(8)))  float    v8f;

#define CAT16(lo, hi) __builtin_shufflevector((lo), (hi), 0,1,2,3,4,5,6,7,8,9,10,11,12,13,14,15)
#define WMMA_F16(a, b, c) __builtin_amdgcn_wmma_f32_16x16x32_f16(false, (a), false, (b), (short)0, (c), false, false)

static constexpr int Bz = 2;
static constexpr int Tz = 2048;
static constexpr int Cz = 1024;
static constexpr int Hz = 16;
static constexpr int HDz = 64;
static constexpr int Mz = Bz * Tz;       // 4096 rows
static constexpr float GSz = 0.1f;

// CDNA5 async global->LDS copy (16B per lane), tracked by ASYNCcnt.
__device__ __forceinline__ void async_b128(unsigned lds_off, const half_t* gptr) {
  asm volatile("global_load_async_to_lds_b128 %0, %1, off"
               :: "v"(lds_off), "v"(gptr) : "memory");
}
__device__ __forceinline__ void wait_async0() {
  asm volatile("s_wait_asynccnt 0" ::: "memory");
}
// Low 32 bits of a generic LDS pointer == wave-relative LDS byte address.
__device__ __forceinline__ unsigned lds_off(const void* p) {
  return (unsigned)(unsigned long long)p;
}

// ---------------------------------------------------------------------------
// fp32 -> f16 conversion (4 elements / thread, counts are multiples of 1024)
// ---------------------------------------------------------------------------
__global__ __launch_bounds__(256) void cvt_f32_to_f16(const float* __restrict__ in,
                                                      half_t* __restrict__ out, int n) {
  int idx = (blockIdx.x * 256 + threadIdx.x) * 4;
  if (idx + 3 < n) {
    float4 f = *(const float4*)(in + idx);
    v4h o;
    o[0] = (half_t)f.x; o[1] = (half_t)f.y; o[2] = (half_t)f.z; o[3] = (half_t)f.w;
    *(v4h*)(out + idx) = o;
  }
}

// ---------------------------------------------------------------------------
// Tiled WMMA GEMM:  D[m,n] = sum_k A[m,k] * W[n,k]
// Block tile 128x128, K-step 64, 8 waves (4x2), wave tile 32x64 = 2x4 frags.
// Double-buffered LDS, async global->LDS staging.
// mode 0: f16 out (+optional f32 bias), row stride ldo
// mode 1: f32 out + bias (final projection), row stride N
// mode 2: guide epilogue: out = t + sigmoid(acc+bias)*adapted*GS  (f16 out)
// ---------------------------------------------------------------------------
__global__ __launch_bounds__(256)
void gemm_nt_wmma(const half_t* __restrict__ A, int lda,
                  const half_t* __restrict__ W, int ldw,
                  int M, int N, int K,
                  const float* __restrict__ bias,
                  int mode,
                  half_t* __restrict__ outh, int ldo,
                  float* __restrict__ outf,
                  const half_t* __restrict__ cat, int ldcat, int catoff,
                  float gs) {
  __shared__ half_t Ash[2][128][72];   // 128 rows x 64 K (+8 pad), x2 buffers
  __shared__ half_t Bsh[2][128][72];

  const int t    = threadIdx.x;
  const int lane = t & 31;
  const int wv   = t >> 5;          // 0..7
  const int wy   = wv >> 1;         // 0..3 -> 32-row strip
  const int wx   = wv & 1;          // 0..1 -> 64-col strip
  const int m0   = blockIdx.y * 128;
  const int n0   = blockIdx.x * 128;

  const int lrow = t >> 1;          // 0..127 staging row
  const int lcol = (t & 1) * 32;    // 0 or 32 halves (64B chunk)

  const int kbA = (lane < 16) ? 0 : 8;    // A-fragment K base per lane half
  const int kbB = (lane < 16) ? 0 : 8;    // B-fragment K base per lane half
  const int ml  = lane & 15;

  const half_t* Arow = A + (size_t)(m0 + lrow) * lda + lcol;
  const half_t* Wrow = W + (size_t)(n0 + lrow) * ldw + lcol;

  auto stage = [&](int buf, int kb) {
    #pragma unroll
    for (int c = 0; c < 4; ++c)
      async_b128(lds_off(&Ash[buf][lrow][lcol + c * 8]), Arow + kb + c * 8);
    #pragma unroll
    for (int c = 0; c < 4; ++c)
      async_b128(lds_off(&Bsh[buf][lrow][lcol + c * 8]), Wrow + kb + c * 8);
  };

  v8f acc[2][4];
  v8f zero = {};
  #pragma unroll
  for (int i = 0; i < 2; ++i)
    #pragma unroll
    for (int j = 0; j < 4; ++j) acc[i][j] = zero;

  stage(0, 0);
  int buf = 0;
  for (int kb = 0; kb < K; kb += 64) {
    wait_async0();          // this wave's tile data resident in LDS
    __syncthreads();        // everyone's tile resident; prev buf fully consumed
    if (kb + 64 < K) stage(buf ^ 1, kb + 64);

    #pragma unroll
    for (int j = 0; j < 2; ++j) {       // two K=32 steps within the 64-tile
      v16h afr[2], bfr[4];
      #pragma unroll
      for (int mi = 0; mi < 2; ++mi) {
        int r = wy * 32 + mi * 16 + ml;
        v8h lo = *(const v8h*)&Ash[buf][r][j * 32 + kbA];
        v8h hi = *(const v8h*)&Ash[buf][r][j * 32 + kbA + 16];
        afr[mi] = CAT16(lo, hi);
      }
      #pragma unroll
      for (int ni = 0; ni < 4; ++ni) {
        int r = wx * 64 + ni * 16 + ml;
        v8h lo = *(const v8h*)&Bsh[buf][r][j * 32 + kbB];
        v8h hi = *(const v8h*)&Bsh[buf][r][j * 32 + kbB + 16];
        bfr[ni] = CAT16(lo, hi);
      }
      #pragma unroll
      for (int ni = 0; ni < 4; ++ni)
        #pragma unroll
        for (int mi = 0; mi < 2; ++mi)
          acc[mi][ni] = WMMA_F16(afr[mi], bfr[ni], acc[mi][ni]);
    }
    buf ^= 1;
  }

  // epilogue: D layout is (VGPR i, lane l) -> m = i + (l<16?0:8), n = l&15
  const int mh = (lane < 16) ? 0 : 8;
  #pragma unroll
  for (int mi = 0; mi < 2; ++mi) {
    #pragma unroll
    for (int ni = 0; ni < 4; ++ni) {
      #pragma unroll
      for (int i = 0; i < 8; ++i) {
        int m = m0 + wy * 32 + mi * 16 + i + mh;
        int n = n0 + wx * 64 + ni * 16 + ml;
        float v = acc[mi][ni][i];
        if (mode == 0) {
          if (bias) v += bias[n];
          outh[(size_t)m * ldo + n] = (half_t)v;
        } else if (mode == 1) {
          outf[(size_t)m * N + n] = v + bias[n];
        } else {  // guide: t + sigmoid(pre)*adapted*GS
          float pre = v + bias[n];
          float g   = 1.0f / (1.0f + __expf(-pre));
          float tt  = (float)cat[(size_t)m * ldcat + n];
          float ad  = (float)cat[(size_t)m * ldcat + catoff + n];
          outh[(size_t)m * ldo + n] = (half_t)(tt + g * ad * gs);
        }
      }
    }
  }
}

// ---------------------------------------------------------------------------
// Flash attention (causal).  Grid: x = T/64 query blocks, y = B*H.
// 128 threads = 4 waves, wave w owns 16 query rows. HD = 64.
// ---------------------------------------------------------------------------
__global__ __launch_bounds__(128)
void flash_attn_wmma(const half_t* __restrict__ Q, const half_t* __restrict__ Kk,
                     const half_t* __restrict__ V, int ldq,
                     half_t* __restrict__ O, int ldo) {
  __shared__ half_t Ksh[64][72];      // [key][d]   (async staged)
  __shared__ half_t Vts[64][72];      // [d][key]   (register-transposed)
  __shared__ half_t Psh[4][16][72];   // per-wave P strip [m][key]

  const int qb   = blockIdx.x;
  const int bh   = blockIdx.y;
  const int b    = bh / Hz;
  const int h    = bh % Hz;
  const int lane = threadIdx.x & 31;
  const int w    = threadIdx.x >> 5;
  const int ml   = lane & 15;
  const int kbA  = (lane < 16) ? 0 : 8;
  const int kbB  = (lane < 16) ? 0 : 8;
  const int mh   = (lane < 16) ? 0 : 8;

  const half_t* Qp = Q + (size_t)(b * Tz) * ldq + h * HDz;
  const half_t* Kp = Kk + (size_t)(b * Tz) * ldq + h * HDz;
  const half_t* Vp = V + (size_t)(b * Tz) * ldq + h * HDz;

  // Q fragments for this wave's 16 rows (A 16x32, two K-steps over d=64)
  v16h aq[2];
  {
    const half_t* qrow = Qp + (size_t)(qb * 64 + w * 16 + ml) * ldq;
    #pragma unroll
    for (int j = 0; j < 2; ++j) {
      v8h lo = *(const v8h*)(qrow + j * 32 + kbA);
      v8h hi = *(const v8h*)(qrow + j * 32 + kbA + 16);
      aq[j] = CAT16(lo, hi);
    }
  }

  float rm[8], rl[8];
  v8f oacc[4];
  v8f zero = {};
  #pragma unroll
  for (int i = 0; i < 8; ++i) { rm[i] = -1e30f; rl[i] = 0.0f; }
  #pragma unroll
  for (int f = 0; f < 4; ++f) oacc[f] = zero;

  const int lrow = threadIdx.x >> 1;       // 0..63
  const int lch  = (threadIdx.x & 1) * 32; // 0 or 32 halves

  for (int kb2 = 0; kb2 <= qb; ++kb2) {
    __syncthreads();   // previous tile fully consumed by all waves
    // K tile: async global->LDS (no layout change needed)
    {
      const half_t* krow = Kp + (size_t)(kb2 * 64 + lrow) * ldq + lch;
      #pragma unroll
      for (int c = 0; c < 4; ++c)
        async_b128(lds_off(&Ksh[lrow][lch + c * 8]), krow + c * 8);
    }
    // V tile: batched loads + register transpose into Vts[d][key]
    {
      const half_t* vrow = Vp + (size_t)(kb2 * 64 + lrow) * ldq + lch;
      v8h vv[4];
      #pragma unroll
      for (int c = 0; c < 4; ++c) vv[c] = *(const v8h*)(vrow + c * 8);
      #pragma unroll
      for (int c = 0; c < 4; ++c)
        #pragma unroll
        for (int e = 0; e < 8; ++e)
          Vts[lch + c * 8 + e][lrow] = vv[c][e];
    }
    wait_async0();
    __syncthreads();

    // S = Q K^T : 4 key-fragments x 2 d-steps
    v8f sacc[4];
    #pragma unroll
    for (int f = 0; f < 4; ++f) sacc[f] = zero;
    #pragma unroll
    for (int j = 0; j < 2; ++j) {
      v16h bfr[4];
      #pragma unroll
      for (int f = 0; f < 4; ++f) {
        int r = f * 16 + ml;                       // key index (B column)
        v8h lo = *(const v8h*)&Ksh[r][j * 32 + kbB];
        v8h hi = *(const v8h*)&Ksh[r][j * 32 + kbB + 16];
        bfr[f] = CAT16(lo, hi);
      }
      #pragma unroll
      for (int f = 0; f < 4; ++f)
        sacc[f] = WMMA_F16(aq[j], bfr[f], sacc[f]);
    }

    // scale + causal mask (diagonal block only)
    const bool diag = (kb2 == qb);
    #pragma unroll
    for (int f = 0; f < 4; ++f) {
      #pragma unroll
      for (int i = 0; i < 8; ++i) {
        float s = sacc[f][i] * 0.125f;             // 1/sqrt(64)
        if (diag && (f * 16 + ml) > (w * 16 + i + mh)) s = -1e30f;
        sacc[f][i] = s;
      }
    }

    // online softmax, per row (VGPR index i <-> row i+mh for this lane half)
    #pragma unroll
    for (int i = 0; i < 8; ++i) {
      float mx = sacc[0][i];
      mx = fmaxf(mx, sacc[1][i]);
      mx = fmaxf(mx, sacc[2][i]);
      mx = fmaxf(mx, sacc[3][i]);
      #pragma unroll
      for (int off = 1; off < 16; off <<= 1)
        mx = fmaxf(mx, __shfl_xor(mx, off, 32));
      float mnew  = fmaxf(rm[i], mx);
      float alpha = __expf(rm[i] - mnew);
      rm[i] = mnew;
      float rs = 0.0f;
      #pragma unroll
      for (int f = 0; f < 4; ++f) {
        float p = __expf(sacc[f][i] - mnew);
        rs += p;
        Psh[w][i + mh][f * 16 + ml] = (half_t)p;
      }
      #pragma unroll
      for (int off = 1; off < 16; off <<= 1)
        rs += __shfl_xor(rs, off, 32);
      rl[i] = rl[i] * alpha + rs;
      #pragma unroll
      for (int f = 0; f < 4; ++f) oacc[f][i] *= alpha;
    }
    __syncthreads();   // publish P (and fence LDS reuse)

    // O += P V : A = P (16 x 64 keys), B = V (keys x d)
    #pragma unroll
    for (int j = 0; j < 2; ++j) {
      v8h plo = *(const v8h*)&Psh[w][ml][j * 32 + kbA];
      v8h phi = *(const v8h*)&Psh[w][ml][j * 32 + kbA + 16];
      v16h pa = CAT16(plo, phi);
      v16h bfr[4];
      #pragma unroll
      for (int f = 0; f < 4; ++f) {
        int r = f * 16 + ml;                       // d column
        v8h lo = *(const v8h*)&Vts[r][j * 32 + kbB];
        v8h hi = *(const v8h*)&Vts[r][j * 32 + kbB + 16];
        bfr[f] = CAT16(lo, hi);
      }
      #pragma unroll
      for (int f = 0; f < 4; ++f)
        oacc[f] = WMMA_F16(pa, bfr[f], oacc[f]);
    }
  }

  // normalize and store (f16, head-interleaved layout (B,T,C))
  #pragma unroll
  for (int f = 0; f < 4; ++f) {
    #pragma unroll
    for (int i = 0; i < 8; ++i) {
      int q = qb * 64 + w * 16 + i + mh;
      int c = h * HDz + f * 16 + ml;
      O[(size_t)(b * Tz + q) * ldo + c] = (half_t)(oacc[f][i] / rl[i]);
    }
  }
}

// ---------------------------------------------------------------------------
// Host-side orchestration
// ---------------------------------------------------------------------------
extern "C" void kernel_launch(void* const* d_in, const int* in_sizes, int n_in,
                              void* d_out, int out_size, void* d_ws, size_t ws_size,
                              hipStream_t stream) {
  const float* x   = (const float*)d_in[0];
  const float* qb  = (const float*)d_in[1];
  const float* kb  = (const float*)d_in[2];
  const float* vb  = (const float*)d_in[3];
  const float* Wq  = (const float*)d_in[4];
  const float* Wk  = (const float*)d_in[5];
  const float* Wv  = (const float*)d_in[6];
  const float* Wqa = (const float*)d_in[7];  const float* bqa = (const float*)d_in[8];
  const float* Wka = (const float*)d_in[9];  const float* bka = (const float*)d_in[10];
  const float* Wva = (const float*)d_in[11]; const float* bva = (const float*)d_in[12];
  const float* Wqg = (const float*)d_in[13]; const float* bqg = (const float*)d_in[14];
  const float* Wkg = (const float*)d_in[15]; const float* bkg = (const float*)d_in[16];
  const float* Wvg = (const float*)d_in[17]; const float* bvg = (const float*)d_in[18];
  const float* Wp  = (const float*)d_in[19]; const float* bp  = (const float*)d_in[20];
  float* out = (float*)d_out;

  // workspace carve-up (all f16 buffers)
  char* base = (char*)d_ws;
  size_t off = 0;
  auto alloc = [&](size_t halves) -> half_t* {
    half_t* p = (half_t*)(base + off);
    off = (off + halves * sizeof(half_t) + 255) & ~(size_t)255;
    return p;
  };
  const size_t MC  = (size_t)Mz * Cz;       // 4096*1024
  const size_t CC  = (size_t)Cz * Cz;       // weight
  const size_t C2C = (size_t)Cz * 2 * Cz;   // gate weight / cat row block

  half_t* xh   = alloc(MC);      // also reused as attention output
  half_t* qbh  = alloc(MC);
  half_t* kbh  = alloc(MC);
  half_t* vbh  = alloc(MC);
  half_t* wqh  = alloc(CC);
  half_t* wkh  = alloc(CC);
  half_t* wvh  = alloc(CC);
  half_t* wqah = alloc(CC);
  half_t* wkah = alloc(CC);
  half_t* wvah = alloc(CC);
  half_t* wph  = alloc(CC);
  half_t* wqgh = alloc(C2C);
  half_t* wkgh = alloc(C2C);
  half_t* wvgh = alloc(C2C);
  half_t* catQ = alloc((size_t)Mz * 2 * Cz);
  half_t* catK = alloc((size_t)Mz * 2 * Cz);
  half_t* catV = alloc((size_t)Mz * 2 * Cz);
  half_t* Qg   = alloc(MC);
  half_t* Kg   = alloc(MC);
  half_t* Vg   = alloc(MC);
  half_t* AO   = xh;  // x no longer needed once QKV projections are done

  auto cvt = [&](const float* src, half_t* dst, size_t n) {
    cvt_f32_to_f16<<<dim3((unsigned)(n / 1024)), 256, 0, stream>>>(src, dst, (int)n);
  };
  cvt(x, xh, MC);  cvt(qb, qbh, MC); cvt(kb, kbh, MC); cvt(vb, vbh, MC);
  cvt(Wq, wqh, CC); cvt(Wk, wkh, CC); cvt(Wv, wvh, CC);
  cvt(Wqa, wqah, CC); cvt(Wka, wkah, CC); cvt(Wva, wvah, CC); cvt(Wp, wph, CC);
  cvt(Wqg, wqgh, C2C); cvt(Wkg, wkgh, C2C); cvt(Wvg, wvgh, C2C);

  dim3 ggrid(Cz / 128, Mz / 128);  // (8, 32)
  auto gemm = [&](const half_t* A, int lda, const half_t* W, int ldw, int K,
                  const float* bias, int mode, half_t* oh, int ldo, float* of,
                  const half_t* cat, int ldcat) {
    gemm_nt_wmma<<<ggrid, 256, 0, stream>>>(A, lda, W, ldw, Mz, Cz, K, bias, mode,
                                            oh, ldo, of, cat, ldcat, Cz, GSz);
  };

  // Stage 1: projections + adapters, writing into concat buffers [t | adapted]
  gemm(xh,  Cz, wqh,  Cz, Cz, nullptr, 0, catQ,      2 * Cz, nullptr, nullptr, 0);
  gemm(xh,  Cz, wkh,  Cz, Cz, nullptr, 0, catK,      2 * Cz, nullptr, nullptr, 0);
  gemm(xh,  Cz, wvh,  Cz, Cz, nullptr, 0, catV,      2 * Cz, nullptr, nullptr, 0);
  gemm(qbh, Cz, wqah, Cz, Cz, bqa,     0, catQ + Cz, 2 * Cz, nullptr, nullptr, 0);
  gemm(kbh, Cz, wkah, Cz, Cz, bka,     0, catK + Cz, 2 * Cz, nullptr, nullptr, 0);
  gemm(vbh, Cz, wvah, Cz, Cz, bva,     0, catV + Cz, 2 * Cz, nullptr, nullptr, 0);

  // Stage 2: gate GEMM (K = 2C) fused with sigmoid guide epilogue
  gemm(catQ, 2 * Cz, wqgh, 2 * Cz, 2 * Cz, bqg, 2, Qg, Cz, nullptr, catQ, 2 * Cz);
  gemm(catK, 2 * Cz, wkgh, 2 * Cz, 2 * Cz, bkg, 2, Kg, Cz, nullptr, catK, 2 * Cz);
  gemm(catV, 2 * Cz, wvgh, 2 * Cz, 2 * Cz, bvg, 2, Vg, Cz, nullptr, catV, 2 * Cz);

  // Stage 3: causal flash attention (grid: 32 q-blocks x 32 (b,h) pairs)
  flash_attn_wmma<<<dim3(Tz / 64, Bz * Hz), 128, 0, stream>>>(Qg, Kg, Vg, Cz, AO, Cz);

  // Stage 4: output projection, f32 result + bias -> d_out
  gemm(AO, Cz, wph, Cz, Cz, bp, 1, nullptr, 0, out, nullptr, 0);

  (void)in_sizes; (void)n_in; (void)out_size; (void)ws_size;
}
